// BlendedMLP_17978733101660
// MI455X (gfx1250) — matrix operand order repaired and checked by reading, not compiled, not run on hardware
//
#include <hip/hip_runtime.h>

typedef __attribute__((ext_vector_type(16))) _Float16 v16h;
typedef __attribute__((ext_vector_type(2)))  __fp16   fp16x2;
typedef __attribute__((ext_vector_type(8)))  float    v8f;

#define NEXP 7

__device__ __forceinline__ float fast_tanh(float v) {
#if __has_builtin(__builtin_amdgcn_tanhf)
    return __builtin_amdgcn_tanhf(v);            // native v_tanh_f32
#else
    float e = __builtin_amdgcn_exp2f(v * 2.8853900817779268f);
    float r = __builtin_amdgcn_rcpf(e + 1.0f);
    return __builtin_fmaf(-2.0f, r, 1.0f);
#endif
}

__device__ __forceinline__ float rfl(float v) {
    return __uint_as_float(__builtin_amdgcn_readfirstlane(__float_as_uint(v)));
}

__device__ __forceinline__ unsigned pack2(float a, float b) {
    fp16x2 pk = __builtin_amdgcn_cvt_pkrtz(a, b);      // v_cvt_pk_rtz_f16_f32
    return __builtin_bit_cast(unsigned, pk);
}

__global__ __launch_bounds__(256, 4) void kan_blend_kernel(
    const float* __restrict__ x,
    const float* __restrict__ knots,
    const float* __restrict__ W1, const float* __restrict__ b1,
    const float* __restrict__ W2, const float* __restrict__ b2,
    const float* __restrict__ W3, const float* __restrict__ b3,
    float* __restrict__ out, int batch)
{
    __shared__ float sW1[NEXP*16], sb1[NEXP*16], sb2[NEXP*16], sW3[NEXP*16], sb3[NEXP];
    const int tid = threadIdx.x;
    if (tid < NEXP*16) { sW1[tid]=W1[tid]; sb1[tid]=b1[tid]; sb2[tid]=b2[tid]; sW3[tid]=W3[tid]; }
    if (tid < NEXP) sb3[tid] = b3[tid];
    __syncthreads();

    const int  lane    = tid & 31;
    const bool lowHalf = lane < 16;
    const int  om      = lane & 15;          // output row M this lane owns in A/C

    // ---- W2 -> WMMA A operand (16x32), this code's contiguous k order: lanes 0-15 K=0..15 in elements 0..15;
    //      lanes 16-31 (K=16..31) are ZERO. This zero padding is
    //      what lets us feed the B operand unmasked: B's lanes 16-31 (K>=16) multiply
    //      against these zeros, so their (finite, tanh-bounded) contents contribute 0.
    v16h A[NEXP];
    #pragma unroll
    for (int n = 0; n < NEXP; ++n) {
        const float* w2 = W2 + n*256 + om*16;
        union AU { v16h h; unsigned u[8]; } a;
        #pragma unroll
        for (int j = 0; j < 8; ++j) a.u[j] = lowHalf ? pack2(w2[2*j], w2[2*j+1]) : 0u;
        A[n] = a.h;
    }

    // ---- knots + safe reciprocals of knot spans (wave-uniform -> SGPRs via readfirstlane)
    float kn[11];
    #pragma unroll
    for (int i = 0; i < 11; ++i) kn[i] = rfl(knots[i]);
    float inv1[10], inv2[9], inv3[8];
    #pragma unroll
    for (int i = 0; i < 10; ++i) { float d = kn[i+1]-kn[i]; inv1[i] = rfl(d != 0.0f ? 1.0f/d : 0.0f); }
    #pragma unroll
    for (int i = 0; i < 9;  ++i) { float d = kn[i+2]-kn[i]; inv2[i] = rfl(d != 0.0f ? 1.0f/d : 0.0f); }
    #pragma unroll
    for (int i = 0; i < 8;  ++i) { float d = kn[i+3]-kn[i]; inv3[i] = rfl(d != 0.0f ? 1.0f/d : 0.0f); }

    const int wavesTotal = (int)(gridDim.x * (blockDim.x >> 5));
    const int waveId     = (int)(blockIdx.x * (blockDim.x >> 5)) + (tid >> 5);

    for (int base = waveId * 32; base < batch; base += wavesTotal * 32) {
        const int  idx   = base + lane;
        const bool valid = idx < batch;
        const float xv   = valid ? x[idx] : 0.0f;

        // ---- cubic B-spline basis (Cox-de Boor, uniform-safe), 7 functions
        float N0[10];
        #pragma unroll
        for (int i = 0; i < 10; ++i)
            N0[i] = ((kn[i] <= xv) && (xv < kn[i+1])) ? 1.0f : 0.0f;
        float N1[9];
        #pragma unroll
        for (int i = 0; i < 9; ++i)
            N1[i] = (xv - kn[i]) * inv1[i] * N0[i] + (kn[i+2] - xv) * inv1[i+1] * N0[i+1];
        float N2[8];
        #pragma unroll
        for (int i = 0; i < 8; ++i)
            N2[i] = (xv - kn[i]) * inv2[i] * N1[i] + (kn[i+3] - xv) * inv2[i+1] * N1[i+1];
        float N3[7];
        #pragma unroll
        for (int i = 0; i < 7; ++i)
            N3[i] = (xv - kn[i]) * inv3[i] * N2[i] + (kn[i+4] - xv) * inv3[i+1] * N2[i+1];

        float acc = 0.0f;
        #pragma unroll
        for (int n = 0; n < NEXP; ++n) {
            // Defeat LICM: re-read the tiny weight vectors from LDS every expert so they
            // don't get hoisted into hundreds of live VGPRs.
            asm volatile("" ::: "memory");

            // ---- layer 1: h1 = tanh(W1*x + b1), packed straight to f16 pairs.
            // Every lane computes its own column; no masking needed (see A padding note).
            union HU { v16h h; unsigned u[8]; };
            HU cur;
            #pragma unroll
            for (int j = 0; j < 8; ++j) {
                float a = fast_tanh(__builtin_fmaf(sW1[n*16 + 2*j],     xv, sb1[n*16 + 2*j]));
                float b = fast_tanh(__builtin_fmaf(sW1[n*16 + 2*j + 1], xv, sb1[n*16 + 2*j + 1]));
                cur.u[j] = pack2(a, b);
            }
            // bring upper-half columns down for the second 16-column tile
            HU oth;
            #pragma unroll
            for (int j = 0; j < 8; ++j) oth.u[j] = __shfl_xor(cur.u[j], 16, 32);

            // layer-2 bias folded into the WMMA accumulator:
            // C row M = r + 8*(upper half)  ==  sb2[n*16 + r + 8*hi]
            const int rb = n*16 + (lowHalf ? 0 : 8);
            v8f Cb;
            #pragma unroll
            for (int r = 0; r < 8; ++r) Cb[r] = sb2[rb + r];

            v8f C0 = __builtin_amdgcn_wmma_f32_16x16x32_f16(false, A[n], false, cur.h,
                                                            (short)0, Cb, false, false);
            v8f C1 = __builtin_amdgcn_wmma_f32_16x16x32_f16(false, A[n], false, oth.h,
                                                            (short)0, Cb, false, false);

            // ---- layer 3: tanh + dot over this lane's 8 rows; b3 seeded as b3/2 per half
            float p0 = 0.5f * sb3[n];
            float p1 = p0;
            #pragma unroll
            for (int r = 0; r < 8; ++r) {
                float w3v = sW3[rb + r];
                p0 = __builtin_fmaf(w3v, fast_tanh(C0[r]), p0);
                p1 = __builtin_fmaf(w3v, fast_tanh(C1[r]), p1);
            }
            // complete 16-row reduction across lane halves (includes both b3/2 seeds)
            float y0 = p0 + __shfl_xor(p0, 16, 32);
            float y1 = p1 + __shfl_xor(p1, 16, 32);
            // tile0 results live on lanes 0-15 (col = lane), tile1 on lanes 16-31 (col = lane)
            acc = __builtin_fmaf(N3[n], lowHalf ? y0 : y1, acc);
        }
        if (valid) out[idx] = acc;
    }
}

extern "C" void kernel_launch(void* const* d_in, const int* in_sizes, int n_in,
                              void* d_out, int out_size, void* d_ws, size_t ws_size,
                              hipStream_t stream) {
    const float* x  = (const float*)d_in[0];
    const float* kn = (const float*)d_in[1];
    const float* W1 = (const float*)d_in[2];
    const float* b1 = (const float*)d_in[3];
    const float* W2 = (const float*)d_in[4];
    const float* b2 = (const float*)d_in[5];
    const float* W3 = (const float*)d_in[6];
    const float* b3 = (const float*)d_in[7];
    float* out = (float*)d_out;
    const int batch = in_sizes[0];

    // 256 threads = 8 waves; each wave covers 32 elements/iter, ~4 iterations/wave
    int blocks = (batch + 1023) / 1024;
    if (blocks < 1) blocks = 1;
    if (blocks > 65535) blocks = 65535;

    kan_blend_kernel<<<blocks, 256, 0, stream>>>(x, kn, W1, b1, W2, b2, W3, b3, out, batch);
}
